// MaskingSelfAttention_23081154249219
// MI455X (gfx1250) — matrix operand order repaired and checked by reading, not compiled, not run on hardware
//
#include <hip/hip_runtime.h>
#include <stdint.h>

typedef __bf16 bf16;
typedef __attribute__((ext_vector_type(16))) __bf16 v16bf;
typedef __attribute__((ext_vector_type(8)))  __bf16 v8bf;
typedef __attribute__((ext_vector_type(8)))  float  v8f;

#define NB   4
#define SEQ  4096
#define DIM  256
#define WPB  8             // waves per block, QKV kernel
#define AWPB 4             // waves per block, attention kernel
#define QROWS (AWPB * 16)  // 64 query rows per attention workgroup

// ---------------------------------------------------------------------------
// fp32 -> bf16 elementwise convert
// ---------------------------------------------------------------------------
__global__ void cvt_bf16_kernel(const float* __restrict__ src,
                                bf16* __restrict__ dst, int n) {
    int i = blockIdx.x * blockDim.x + threadIdx.x;
    int stride = gridDim.x * blockDim.x;
    for (; i < n; i += stride) dst[i] = (bf16)src[i];
}

// ---------------------------------------------------------------------------
// CDNA5 async global -> LDS copy (16B per lane), tracked by ASYNCcnt
// ---------------------------------------------------------------------------
__device__ __forceinline__ void async_ld_b128(uint32_t lds_byte, const void* g) {
    asm volatile("global_load_async_to_lds_b128 %0, %1, off"
                 :: "v"(lds_byte), "v"((uint64_t)(uintptr_t)g)
                 : "memory");
}
__device__ __forceinline__ void wait_async0() {
    asm volatile("s_wait_asynccnt 0" ::: "memory");
}
__device__ __forceinline__ void wait_async8() {
    asm volatile("s_wait_asynccnt 8" ::: "memory");
}
__device__ __forceinline__ uint32_t lds_off(const void* p) {
    return (uint32_t)(uintptr_t)p;   // LDS aperture lives in the high word
}

// ---------------------------------------------------------------------------
// WMMA fragment loaders (wave32, 16x16x32 bf16; ISA 7.12.2 layouts)
// ---------------------------------------------------------------------------
union ABFrag { v16bf v; v8bf h[2]; };

// A (16x32, MxK): lane = (m = lane&15, half = lane>>4)
//   regs 0..3 hold K = 8*half .. 8*half+7, regs 4..7 hold K = 16+8*half .. +7
__device__ __forceinline__ v16bf load_frag_a(const bf16* base, int ld, int lane) {
    int m = lane & 15, hf = lane >> 4;
    ABFrag u;
    u.h[0] = *(const v8bf*)(base + (size_t)m * ld + 8 * hf);
    u.h[1] = *(const v8bf*)(base + (size_t)m * ld + 16 + 8 * hf);
    return u.v;
}

// B (32x16, KxN) sourced from S[n][k] (k contiguous):
//   lane = (n = lane&15, half = lane>>4), regs 0..3 hold K = 8*half .. +7, regs 4..7 hold K = 16+8*half .. +7
__device__ __forceinline__ v16bf load_frag_b_nk(const bf16* base, int ld, int lane) {
    int n = lane & 15, hf = lane >> 4;
    ABFrag u;
    u.h[0] = *(const v8bf*)(base + (size_t)n * ld + 8 * hf);
    u.h[1] = *(const v8bf*)(base + (size_t)n * ld + 16 + 8 * hf);
    return u.v;
}

// ---------------------------------------------------------------------------
// QKV projection: y = x @ W.T + b, one wave per 16x16 output tile.
// which==0 -> q (row major), which==1 -> k (row major), which==2 -> vT[e][tok]
// ---------------------------------------------------------------------------
__global__ void __launch_bounds__(256)
qkv_kernel(const bf16* __restrict__ xb,
           const bf16* __restrict__ wq, const bf16* __restrict__ wk,
           const bf16* __restrict__ wv,
           const float* __restrict__ biasq, const float* __restrict__ biask,
           const float* __restrict__ biasv,
           bf16* __restrict__ qout, bf16* __restrict__ kout,
           bf16* __restrict__ vTout) {
    int lane = threadIdx.x & 31;
    int wave = threadIdx.x >> 5;
    long task = (long)blockIdx.x * WPB + wave;   // 1024 row-tiles * 16 col-tiles * 3
    int which = (int)(task % 3);
    long t2 = task / 3;
    int ct = (int)(t2 & 15);        // output-column tile (e)
    int rt = (int)(t2 >> 4);        // row tile (token)
    if (rt >= (NB * SEQ) / 16) return;

    const bf16* W = (which == 0) ? wq : (which == 1) ? wk : wv;
    const float* bias = (which == 0) ? biasq : (which == 1) ? biask : biasv;

    int m0 = rt * 16, n0 = ct * 16;
    v8f acc = {0.f, 0.f, 0.f, 0.f, 0.f, 0.f, 0.f, 0.f};
#pragma unroll
    for (int kt = 0; kt < 8; kt++) {
        v16bf a = load_frag_a(xb + (size_t)m0 * DIM + kt * 32, DIM, lane);
        v16bf b = load_frag_b_nk(W + (size_t)n0 * DIM + kt * 32, DIM, lane);
        acc = __builtin_amdgcn_wmma_f32_16x16x32_bf16(false, a, false, b,
                                                      (short)0, acc, false, false);
    }
    int nl = lane & 15, hf = lane >> 4;
    float bval = bias[n0 + nl];
    if (which < 2) {
        bf16* dst = (which == 0) ? qout : kout;
#pragma unroll
        for (int r = 0; r < 8; r++) {
            int g = m0 + r + 8 * hf;                    // global token row
            dst[(size_t)g * DIM + n0 + nl] = (bf16)(acc[r] + bval);
        }
    } else {
#pragma unroll
        for (int r = 0; r < 8; r++) {
            int g = m0 + r + 8 * hf;
            int bb = g >> 12;                           // / SEQ
            int tok = g & (SEQ - 1);
            vTout[(size_t)bb * DIM * SEQ + (size_t)(n0 + nl) * SEQ + tok] =
                (bf16)(acc[r] + bval);
        }
    }
}

// ---------------------------------------------------------------------------
// Causal flash attention. Block = 4 waves = 64 query rows of one batch.
// Double-buffered async K staging + async V staging overlapped with WMMA.
// ---------------------------------------------------------------------------
__global__ void __launch_bounds__(128)
attn_kernel(const bf16* __restrict__ qmat, const bf16* __restrict__ kmat,
            const bf16* __restrict__ vT, float* __restrict__ out) {
    __shared__ bf16 Kt[2][32 * DIM];       // [buf][key][d]  32 KB
    __shared__ bf16 Vt[DIM * 32];          // [e][key]       16 KB
    __shared__ bf16 Pt[AWPB][16 * 32];     // P transpose bounce, 4 KB

    int lane = threadIdx.x & 31;
    int wave = threadIdx.x >> 5;
    int blk = blockIdx.x;                         // 256 blocks total
    int b = blk >> 6;                             // SEQ/QROWS = 64 blocks/batch
    int q0_blk = (blk & 63) * QROWS;              // block's first query row
    int q0 = q0_blk + wave * 16;                  // wave's first query row
    int n_keys = q0_blk + QROWS;                  // keys needed by whole block
    int my_kmax = q0 + 15;                        // wave's last query row
    int ntiles = n_keys >> 5;                     // 32-key tiles

    const bf16* qb = qmat + (size_t)b * SEQ * DIM;
    const bf16* kb = kmat + (size_t)b * SEQ * DIM;
    const bf16* vtb = vT + (size_t)b * DIM * SEQ;

    int nl = lane & 15, hf = lane >> 4;

    uint32_t kt_base[2] = { lds_off(&Kt[0][0]), lds_off(&Kt[1][0]) };
    uint32_t vt_base = lds_off(&Vt[0]);

    // Q fragments for this wave's 16 rows, all of D (8 k-tiles of 32)
    v16bf qf[8];
#pragma unroll
    for (int t = 0; t < 8; t++)
        qf[t] = load_frag_a(qb + (size_t)q0 * DIM + t * 32, DIM, lane);

    const v8f zf = {0.f, 0.f, 0.f, 0.f, 0.f, 0.f, 0.f, 0.f};
    v8f o[16];
#pragma unroll
    for (int t = 0; t < 16; t++) o[t] = zf;
    float mrow[8], lrow[8];
#pragma unroll
    for (int r = 0; r < 8; r++) { mrow[r] = -__builtin_inff(); lrow[r] = 0.f; }

    // prologue: start K tile 0 (8 x b128 per thread, 32x256 bf16 contiguous)
#pragma unroll
    for (int it = 0; it < 8; it++) {
        int j = threadIdx.x + it * 128;
        async_ld_b128(kt_base[0] + j * 16, kb + j * 8);
    }

    for (int t = 0; t < ntiles; t++) {
        int kb0 = t * 32;
        int p = t & 1;
        bool has_next = (t + 1 < ntiles);
        bool active = (kb0 <= my_kmax);           // wave-uniform

        wait_async0();      // own K[t] chunks landed
        __syncthreads();    // everyone's K[t] landed; V[t-1]/old K consumed

        // issue V[t]: row e -> 32 contiguous keys (4 x b128), oldest in queue
#pragma unroll
        for (int rr = 0; rr < 2; rr++) {
            int e = threadIdx.x + rr * 128;
            const bf16* vs = vtb + (size_t)e * SEQ + kb0;
#pragma unroll
            for (int c = 0; c < 4; c++)
                async_ld_b128(vt_base + e * 64 + c * 16, vs + c * 8);
        }
        // issue K[t+1] into alternate buffer (8 x b128), youngest in queue
        if (has_next) {
            const bf16* ksrc = kb + (size_t)(kb0 + 32) * DIM;
#pragma unroll
            for (int it = 0; it < 8; it++) {
                int j = threadIdx.x + it * 128;
                async_ld_b128(kt_base[p ^ 1] + j * 16, ksrc + j * 8);
            }
        }

        v16bf pf;
        if (active) {
            const bf16* Kcur = Kt[p];
            // S = Q Kt^T for keys [kb0, kb0+32): two 16-wide subtiles
            v8f s0 = zf, s1 = zf;
#pragma unroll
            for (int dt = 0; dt < 8; dt++) {
                v16bf k0 = load_frag_b_nk(Kcur + 0 * DIM + dt * 32, DIM, lane);
                s0 = __builtin_amdgcn_wmma_f32_16x16x32_bf16(false, qf[dt], false, k0,
                                                             (short)0, s0, false, false);
                v16bf k1 = load_frag_b_nk(Kcur + 16 * DIM + dt * 32, DIM, lane);
                s1 = __builtin_amdgcn_wmma_f32_16x16x32_bf16(false, qf[dt], false, k1,
                                                             (short)0, s1, false, false);
            }

            // causal mask (C layout: col = nl, row = r + 8*hf)
#pragma unroll
            for (int r = 0; r < 8; r++) {
                int mg = q0 + r + 8 * hf;
                if (kb0 + nl > mg)      s0[r] = -__builtin_inff();
                if (kb0 + 16 + nl > mg) s1[r] = -__builtin_inff();
            }

            // online softmax: row stats live per (reg r, half) slot
            float scl[8];
#pragma unroll
            for (int r = 0; r < 8; r++) {
                float v = fmaxf(s0[r], s1[r]);
                v = fmaxf(v, __shfl_xor(v, 1, 32));
                v = fmaxf(v, __shfl_xor(v, 2, 32));
                v = fmaxf(v, __shfl_xor(v, 4, 32));
                v = fmaxf(v, __shfl_xor(v, 8, 32));
                float mn = fmaxf(mrow[r], v);
                float p0 = __expf(s0[r] - mn);
                float p1 = __expf(s1[r] - mn);
                s0[r] = p0; s1[r] = p1;
                float rs = p0 + p1;
                rs += __shfl_xor(rs, 1, 32);
                rs += __shfl_xor(rs, 2, 32);
                rs += __shfl_xor(rs, 4, 32);
                rs += __shfl_xor(rs, 8, 32);
                float sc = __expf(mrow[r] - mn);
                lrow[r] = lrow[r] * sc + rs;
                mrow[r] = mn;
                scl[r] = sc;
            }
#pragma unroll
            for (int et = 0; et < 16; et++)
#pragma unroll
                for (int r = 0; r < 8; r++) o[et][r] *= scl[r];

            // C-layout -> A-layout transpose of P through per-wave LDS
            bf16* P = Pt[wave];
#pragma unroll
            for (int r = 0; r < 8; r++) {
                P[(r + 8 * hf) * 32 + nl]      = (bf16)s0[r];
                P[(r + 8 * hf) * 32 + 16 + nl] = (bf16)s1[r];
            }
            asm volatile("" ::: "memory");   // keep DS store->load order
            pf = load_frag_a(P, 32, lane);
        }

        // V[t] landed (leave K[t+1]'s 8 loads in flight when present)
        if (has_next) wait_async8(); else wait_async0();
        __syncthreads();    // everyone's V[t] landed

        if (active) {
            // O += P @ V  (16 e-tiles, K=32)
#pragma unroll
            for (int et = 0; et < 16; et++) {
                v16bf vf = load_frag_b_nk(Vt + et * 16 * 32, 32, lane);
                o[et] = __builtin_amdgcn_wmma_f32_16x16x32_bf16(false, pf, false, vf,
                                                                (short)0, o[et], false, false);
            }
        }
    }

    // epilogue: O / l -> fp32 out
    float* ob = out + ((size_t)b * SEQ + q0) * DIM;
#pragma unroll
    for (int r = 0; r < 8; r++) {
        float inv = 1.0f / lrow[r];
#pragma unroll
        for (int t = 0; t < 16; t++)
            ob[(size_t)(r + 8 * hf) * DIM + t * 16 + nl] = o[t][r] * inv;
    }
}

// ---------------------------------------------------------------------------
// launch
// ---------------------------------------------------------------------------
extern "C" void kernel_launch(void* const* d_in, const int* in_sizes, int n_in,
                              void* d_out, int out_size, void* d_ws, size_t ws_size,
                              hipStream_t stream) {
    const float* x  = (const float*)d_in[0];
    const float* Wq = (const float*)d_in[1];
    const float* bq = (const float*)d_in[2];
    const float* Wk = (const float*)d_in[3];
    const float* bk = (const float*)d_in[4];
    const float* Wv = (const float*)d_in[5];
    const float* bv = (const float*)d_in[6];
    float* out = (float*)d_out;

    const size_t NTOK = (size_t)NB * SEQ;        // 16384
    const size_t XSZ  = NTOK * DIM;              // 4 Mi elems
    const size_t WSZ  = (size_t)DIM * DIM;       // 64 Ki elems

    char* ws = (char*)d_ws;
    bf16* xb  = (bf16*)(ws);                              // 8 MB
    bf16* qb  = (bf16*)(ws + XSZ * 2);                    // 8 MB
    bf16* kbf = (bf16*)(ws + XSZ * 4);                    // 8 MB
    bf16* vTb = (bf16*)(ws + XSZ * 6);                    // 8 MB
    bf16* wqb = (bf16*)(ws + XSZ * 8);                    // 128 KB
    bf16* wkb = (bf16*)(ws + XSZ * 8 + WSZ * 2);
    bf16* wvb = (bf16*)(ws + XSZ * 8 + WSZ * 4);

    cvt_bf16_kernel<<<4096, 256, 0, stream>>>(x,  xb,  (int)XSZ);
    cvt_bf16_kernel<<<256,  256, 0, stream>>>(Wq, wqb, (int)WSZ);
    cvt_bf16_kernel<<<256,  256, 0, stream>>>(Wk, wkb, (int)WSZ);
    cvt_bf16_kernel<<<256,  256, 0, stream>>>(Wv, wvb, (int)WSZ);

    // 1024 row-tiles * 16 col-tiles * 3 matrices = 49152 wave-tasks, 8 per block
    qkv_kernel<<<49152 / WPB, 256, 0, stream>>>(xb, wqb, wkb, wvb, bq, bk, bv,
                                                qb, kbf, vTb);

    // 4 batches * (4096/64) = 256 blocks of 128 threads
    attn_kernel<<<(NB * SEQ) / QROWS, 128, 0, stream>>>(qb, kbf, vTb, out);
}
